// EfficientInteractionBilinear_78408922956492
// MI455X (gfx1250) — matrix operand order repaired and checked by reading, not compiled, not run on hardware
//
#include <hip/hip_runtime.h>
#include <hip/hip_fp16.h>

typedef __attribute__((ext_vector_type(16))) _Float16 v16h;
typedef __attribute__((ext_vector_type(8)))  _Float16 v8h;
typedef __attribute__((ext_vector_type(8)))  float    v8f;

#define N_SPH   7
#define KMAX    16
#define EMB     128
#define INTERM  64
#define UNITS   128
#define KTOT    (INTERM * EMB)     // 8192  (GEMM K for step 3)
#define KSTEP   32                 // K per v_wmma_f32_16x16x32_f16
#define NKS     (KTOT / KSTEP)     // 256 k-steps
#define NTILES  (UNITS / 16)       // 8 N-tiles -> 8 waves
#define EPB     16                 // edges per block (GEMM M)

// LDS layout (bytes):
//   [0, 262144)          tmpA : f16 [16 edges][8192]   (A operand for step-3 GEMM)
//   [0, 131072)          m2f  : f32 [16][16][128]      (overlaps tmpA; dead before tmpA written)
//   [262144, 319488)     sumk : f32 [16][7][128]
#define TMPA_BYTES (EPB * KTOT * 2)             // 262144
#define SUMK_OFF   TMPA_BYTES
#define SUMK_ELTS  (EPB * N_SPH * EMB)          // 14336 floats
#define LDS_TOTAL  (TMPA_BYTES + SUMK_ELTS * 4) // 319488 <= 320KB WGP LDS

// ---------------------------------------------------------------------------
// Kernel 0: repack weight[em][i][o] (f32) into f16 WMMA-B fragments.
// w2p index = ((ks*NTILES + nt)*32 + lane)*16 + h ; B element (K,N) with
//   N = nt*16 + (lane&15),  K = ks*32 + (lane>>4)*16 + h   (dense 32x16 B map)
//   K -> (i, em) via K = i*128 + em, matching tmpA's column order.
// ---------------------------------------------------------------------------
__global__ __launch_bounds__(256)
void eib_pack_weight(const float* __restrict__ w, _Float16* __restrict__ w2p) {
  int idx = blockIdx.x * blockDim.x + threadIdx.x;
  if (idx >= NKS * NTILES * 32 * 16) return;
  int h    = idx & 15;
  int lane = (idx >> 4) & 31;
  int nt   = (idx >> 9) & (NTILES - 1);
  int ks   = idx >> 12;
  int g    = lane >> 4;
  int o    = nt * 16 + (lane & 15);
  int k    = ks * KSTEP + g * 16 + h;
  int i    = k >> 7;        // /128
  int em   = k & 127;
  w2p[idx] = (_Float16)w[(size_t)em * (INTERM * UNITS) + (size_t)i * UNITS + o];
}

// ---------------------------------------------------------------------------
// Kernel 1: fused scatter + (sph@m2) + (rbf@sumk) + bilinear WMMA GEMM.
// ---------------------------------------------------------------------------
__global__ __launch_bounds__(256)
void eib_fused(const float* __restrict__ rbf,   // [nE][64][7]
               const float* __restrict__ sph,   // [nE][7][16]
               const float* __restrict__ m,     // [nE*16][128]
               const int*   __restrict__ id_reduce,
               const int*   __restrict__ kidx,
               const _Float16* __restrict__ w2p, // packed B, 2MB (L2-resident)
               float* __restrict__ out,          // [nE][128]
               int nEdges) {
  extern __shared__ char smem[];
  float*    m2f  = (float*)smem;                 // [16][16][128] f32
  _Float16* tmpA = (_Float16*)smem;              // [16][8192]    f16
  float*    sumk = (float*)(smem + SUMK_OFF);    // [16][7][128]  f32

  const int tid   = threadIdx.x;
  const int edge0 = blockIdx.x * EPB;

  // --- phase 0a: zero m2 tile (scatter target) ---
  for (int i = tid; i < EPB * KMAX * EMB; i += 256) m2f[i] = 0.f;
  __syncthreads();

  // --- phase 0b: scatter triplet embeddings using the index arrays.
  // Each edge owns exactly KMAX triplet slots, so this block's 256 triplets
  // are t in [blk*256, blk*256+256).
  {
    long long t = (long long)blockIdx.x * (EPB * KMAX) + tid;
    int eloc = id_reduce[t] - edge0;       // 0..15
    int k    = kidx[t];                    // 0..15
    if (eloc >= 0 && eloc < EPB && k >= 0 && k < KMAX) {
      const float4* src = (const float4*)(m + t * EMB);
      float4* dst = (float4*)(m2f + (eloc * KMAX + k) * EMB);
      #pragma unroll
      for (int j = 0; j < EMB / 4; ++j) dst[j] = src[j];
    }
  }
  __syncthreads();

  // --- phase 1: sumk[e][s][em] = sum_k sph[e][s][k] * m2[e][k][em]  (fp32) ---
  for (int idx = tid; idx < SUMK_ELTS; idx += 256) {
    int e  = idx / (N_SPH * EMB);
    int r  = idx - e * (N_SPH * EMB);
    int s  = r / EMB;
    int em = r - s * EMB;
    const float* sp = sph + ((size_t)(edge0 + e) * N_SPH + s) * KMAX;
    const float* mm = m2f + (e * KMAX) * EMB + em;
    float acc = 0.f;
    #pragma unroll
    for (int k = 0; k < KMAX; ++k) acc += sp[k] * mm[k * EMB];
    sumk[idx] = acc;
  }
  __syncthreads();   // m2f now dead; tmpA may be written

  // --- phase 2: tmpA[e][i*128+em] = f16( sum_s rbf[e][i][s] * sumk[e][s][em] )
  #pragma unroll
  for (int p = 0; p < 4; ++p) {
    int pidx = tid + p * 256;        // 0..1023 = (e,i) pairs
    int e = pidx >> 6;
    int i = pidx & 63;
    const float* rb = rbf + ((size_t)(edge0 + e) * INTERM + i) * N_SPH;
    float r0 = rb[0], r1 = rb[1], r2 = rb[2], r3 = rb[3],
          r4 = rb[4], r5 = rb[5], r6 = rb[6];
    const float* sk = sumk + e * (N_SPH * EMB);
    _Float16* dst = tmpA + (size_t)e * KTOT + i * EMB;
    for (int em = 0; em < EMB; ++em) {
      float acc = r0 * sk[em]           + r1 * sk[EMB + em]
                + r2 * sk[2 * EMB + em] + r3 * sk[3 * EMB + em]
                + r4 * sk[4 * EMB + em] + r5 * sk[5 * EMB + em]
                + r6 * sk[6 * EMB + em];
      dst[em] = (_Float16)acc;
    }
  }
  __syncthreads();

  // --- phase 3: out_tile(16x128) = tmpA(16x8192) @ W2(8192x128) via WMMA ---
  // Software-pipelined: fragments for k-step ks+1 are loaded before the WMMA
  // for ks issues, so the pre-WMMA wait is partial (overlap load & matrix op).
  // wave w owns N-tile w; lane: M = lane&15, group g = lane>>4.
  // A fragment (contiguous 16x32 layout, matching w2p): half h -> K = 16g + h
  //   => two aligned 16B LDS reads at K-offsets {16g, 16g+8}.
  const int wave = tid >> 5;
  const int lane = tid & 31;
  const int g    = lane >> 4;
  const int mrow = lane & 15;
  const _Float16* arow = tmpA + (size_t)mrow * KTOT;
  const _Float16* bcol = w2p + ((size_t)wave * 32 + lane) * 16;
  const size_t bstride = (size_t)NTILES * 32 * 16;   // halves per k-step

  auto load_a = [&](int ks) -> v16h {
    const v8h lo = *(const v8h*)(arow + ks * KSTEP + 16 * g);
    const v8h hi = *(const v8h*)(arow + ks * KSTEP + 16 * g + 8);
    v16h a;
    #pragma unroll
    for (int j = 0; j < 8; ++j) { a[j] = lo[j]; a[8 + j] = hi[j]; }
    return a;
  };

  v8f c = {};
  v16h a0 = load_a(0);
  v16h b0 = *(const v16h*)(bcol);
  #pragma unroll 4
  for (int ks = 0; ks < NKS - 1; ++ks) {
    int pks = (ks + 8 < NKS) ? ks + 8 : NKS - 1;     // clamp: branch-free
    __builtin_prefetch(bcol + (size_t)pks * bstride, 0, 1);
    v16h a1 = load_a(ks + 1);
    v16h b1 = *(const v16h*)(bcol + (size_t)(ks + 1) * bstride);
    c = __builtin_amdgcn_wmma_f32_16x16x32_f16(
            /*neg_a=*/false, a0, /*neg_b=*/false, b0,
            /*c_mod=*/(short)0, c, /*reuse_a=*/false, /*reuse_b=*/false);
    a0 = a1; b0 = b1;
  }
  c = __builtin_amdgcn_wmma_f32_16x16x32_f16(
          false, a0, false, b0, (short)0, c, false, false);

  // C layout: VGPR r -> M = r + 8g, N = mrow.
  #pragma unroll
  for (int r = 0; r < 8; ++r) {
    int e = edge0 + r + 8 * g;
    if (e < nEdges) out[(size_t)e * UNITS + wave * 16 + mrow] = c[r];
  }
}

extern "C" void kernel_launch(void* const* d_in, const int* in_sizes, int n_in,
                              void* d_out, int out_size, void* d_ws, size_t ws_size,
                              hipStream_t stream) {
  const float* rbf = (const float*)d_in[0];   // (nE, 64, 7)
  const float* sph = (const float*)d_in[1];   // (nE, 7, 16)
  const float* m   = (const float*)d_in[2];   // (nE*16, 128)
  const float* w   = (const float*)d_in[3];   // (128, 64, 128)
  const int* idr   = (const int*)d_in[4];
  const int* kx    = (const int*)d_in[5];
  float* out       = (float*)d_out;
  _Float16* w2p    = (_Float16*)d_ws;         // 2 MB packed f16 B matrix

  const int nEdges = in_sizes[0] / (INTERM * N_SPH);

  const int packElts = NKS * NTILES * 32 * 16;          // 1,048,576
  eib_pack_weight<<<(packElts + 255) / 256, 256, 0, stream>>>(w, w2p);

  const int nBlocks = (nEdges + EPB - 1) / EPB;         // 1250
  eib_fused<<<nBlocks, 256, LDS_TOTAL, stream>>>(rbf, sph, m, idr, kx,
                                                 w2p, out, nEdges);
}
